// ResamplerAttention_32933809225997
// MI455X (gfx1250) — compile-verified, matrix operand order checked by reading, not run on hardware
//
#include <hip/hip_runtime.h>
#include <math.h>

// ---------------------------------------------------------------------------
// Types for CDNA5 WMMA (wave32, 16x16x32 bf16 -> f32)
// ---------------------------------------------------------------------------
typedef __attribute__((ext_vector_type(16))) __bf16 v16bf;
typedef __attribute__((ext_vector_type(8)))  __bf16 v8bf;
typedef __attribute__((ext_vector_type(4)))  __bf16 v4bf;
typedef __attribute__((ext_vector_type(8)))  float  v8f;
typedef __attribute__((ext_vector_type(4)))  unsigned int u32x4;
typedef __attribute__((ext_vector_type(8)))  unsigned int u32x8;

#define ATT_SCALE 0.125f           // HEAD_DIM^-0.5, HEAD_DIM = 64
#define MASK_FILL -1000000000.0f

static __device__ __forceinline__ v8f wmma_bf16(v16bf a, v16bf b, v8f c) {
  return __builtin_amdgcn_wmma_f32_16x16x32_bf16(
      /*neg_a=*/false, a, /*neg_b=*/false, b,
      /*c_mod=*/(short)0, c, /*reuse_a=*/false, /*reuse_b=*/false);
}

// Load the per-lane element set of a 16x32 bf16 A-fragment (or B-fragment when
// B is stored transposed / N-major) from a row pointer.
// CDNA5 16-bit A layout (ISA 7.12.2): lane L holds row L%16; lanes 0-15 cover
// K = kb+{0..7, 16..23}, lanes 16-31 cover K = kb+{8..15, 24..31}.
static __device__ __forceinline__ v16bf load_frag(const __bf16* row, int kb, int lane) {
  const __bf16* p = row + kb + ((lane >= 16) ? 8 : 0);
  v8bf lo = *(const v8bf*)(p);        // K = base .. base+7
  v8bf hi = *(const v8bf*)(p + 16);   // K = base+16 .. base+23
  v16bf r;
#pragma unroll
  for (int i = 0; i < 8; ++i) { r[i] = lo[i]; r[i + 8] = hi[i]; }
  return r;
}

// ---------------------------------------------------------------------------
// Tensor Data Mover: 2D tile load (rows x cols of bf16) from global to LDS.
// D# built per CDNA5 ISA ch.8: group0 = {ctrl, lds_addr, global_addr, type=2},
// group1 = {mask/data_size, tensor dims, tile dims, strides}.  Issued once per
// wave (EXEC-independent); completion via TENSORcnt.
// ---------------------------------------------------------------------------
static __device__ __forceinline__ void tdm_load_2d(unsigned lds_off, unsigned long long gaddr,
                                                   unsigned tensor_d0, unsigned tensor_d1,
                                                   unsigned tile_d0, unsigned tile_d1,
                                                   unsigned stride0) {
  u32x4 g0;
  g0[0] = 1u;                                              // count=1, user mode
  g0[1] = lds_off;                                         // lds_addr (bytes)
  g0[2] = (unsigned)gaddr;                                 // global_addr[31:0]
  g0[3] = ((unsigned)(gaddr >> 32) & 0x1FFFFFFu) | (2u << 30);  // ga[56:32] | type=2
  u32x8 g1;
  g1[0] = 1u << 16;                                        // wg_mask=0, data_size=2B
  g1[1] = (tensor_d0 & 0xFFFFu) << 16;                     // barrier_addr=0 | td0 lo
  g1[2] = (tensor_d0 >> 16) | ((tensor_d1 & 0xFFFFu) << 16);   // td0 hi | td1 lo
  g1[3] = (tensor_d1 >> 16) | (tile_d0 << 16);             // td1 hi | tile_dim0
  g1[4] = tile_d1;                                         // tile_dim1 | tile_dim2=0
  g1[5] = stride0;                                         // tensor_dim0_stride lo
  g1[6] = 0u;                                              // stride0 hi | stride1 lo
  g1[7] = 0u;
  asm volatile("tensor_load_to_lds %0, %1" :: "s"(g0), "s"(g1) : "memory");
}

// ---------------------------------------------------------------------------
// LayerNorm (fp32 in) fused with bf16 conversion.  One 256-thread block / row,
// row length fixed at 1024.
// ---------------------------------------------------------------------------
__global__ void layernorm_to_bf16(const float* __restrict__ in,
                                  const float* __restrict__ gamma,
                                  const float* __restrict__ beta,
                                  __bf16* __restrict__ out) {
  const int row = blockIdx.x;
  const float4 v = ((const float4*)(in + (size_t)row * 1024))[threadIdx.x];
  float s  = v.x + v.y + v.z + v.w;
  float sq = v.x * v.x + v.y * v.y + v.z * v.z + v.w * v.w;
#pragma unroll
  for (int off = 16; off; off >>= 1) {
    s  += __shfl_xor(s,  off, 32);
    sq += __shfl_xor(sq, off, 32);
  }
  __shared__ float ws[8], wq[8];
  const int wv = threadIdx.x >> 5, ln = threadIdx.x & 31;
  if (ln == 0) { ws[wv] = s; wq[wv] = sq; }
  __syncthreads();
  float ts = 0.f, tq = 0.f;
#pragma unroll
  for (int i = 0; i < 8; ++i) { ts += ws[i]; tq += wq[i]; }
  const float mu   = ts * (1.0f / 1024.0f);
  const float var  = tq * (1.0f / 1024.0f) - mu * mu;
  const float rstd = rsqrtf(var + 1e-5f);

  const float4 g = ((const float4*)gamma)[threadIdx.x];
  const float4 b = ((const float4*)beta)[threadIdx.x];
  v4bf o;
  o[0] = (__bf16)((v.x - mu) * rstd * g.x + b.x);
  o[1] = (__bf16)((v.y - mu) * rstd * g.y + b.y);
  o[2] = (__bf16)((v.z - mu) * rstd * g.z + b.z);
  o[3] = (__bf16)((v.w - mu) * rstd * g.w + b.w);
  *(v4bf*)(out + (size_t)row * 1024 + threadIdx.x * 4) = o;
}

// ---------------------------------------------------------------------------
// fp32 [K][N] -> bf16 transposed [N][K] (LDS 32x32 tile, block (32,8))
// ---------------------------------------------------------------------------
__global__ void transpose_to_bf16(const float* __restrict__ in,
                                  __bf16* __restrict__ out, int K, int N) {
  __shared__ float tile[32][33];
  const int k0 = blockIdx.y * 32, n0 = blockIdx.x * 32;
  const int tx = threadIdx.x, ty = threadIdx.y;
#pragma unroll
  for (int r = ty; r < 32; r += 8)
    tile[r][tx] = in[(size_t)(k0 + r) * N + (n0 + tx)];
  __syncthreads();
#pragma unroll
  for (int r = ty; r < 32; r += 8)
    out[(size_t)(n0 + r) * K + (k0 + tx)] = (__bf16)tile[tx][r];
}

// ---------------------------------------------------------------------------
// Wave-tile GEMM core (global B): one wave produces a 32(M) x 64(N) f32 tile.
// A: [M][K] bf16 row-major.  BT: [N][K] bf16 (B transposed).  2 A-fragments
// x 4 B-fragments -> 8 WMMAs per K-step (B reused x2, A reused x4).
// ---------------------------------------------------------------------------
struct Acc8 { v8f r0[4]; v8f r1[4]; };

static __device__ __forceinline__ Acc8 gemm_wave_32x64(
    const __bf16* __restrict__ A, const __bf16* __restrict__ BT,
    int lda, int ldb, int K, int m0, int n0, int lane) {
  const int lm = lane & 15;
  const __bf16* arow0 = A + (size_t)(m0 + lm) * lda;
  const __bf16* arow1 = arow0 + (size_t)16 * lda;
  const __bf16* br[4];
#pragma unroll
  for (int t = 0; t < 4; ++t) br[t] = BT + (size_t)(n0 + t * 16 + lm) * ldb;
  Acc8 acc = {};
  for (int kb = 0; kb < K; kb += 32) {
    __builtin_prefetch(arow0 + kb + 128, 0, 0);
    __builtin_prefetch(arow1 + kb + 128, 0, 0);
    v16bf a0 = load_frag(arow0, kb, lane);
    v16bf a1 = load_frag(arow1, kb, lane);
#pragma unroll
    for (int t = 0; t < 4; ++t) {
      v16bf bt = load_frag(br[t], kb, lane);
      acc.r0[t] = wmma_bf16(a0, bt, acc.r0[t]);
      acc.r1[t] = wmma_bf16(a1, bt, acc.r1[t]);
    }
  }
  return acc;
}

// C/D f32 16x16 layout: VGPR j, lane L -> row j + (L>=16 ? 8 : 0), col L%16.

// Plain GEMM -> bf16 row-major C  (used for Q projection)
__global__ void gemm_bf16_kernel(const __bf16* __restrict__ A,
                                 const __bf16* __restrict__ BT,
                                 __bf16* __restrict__ C, int K, int N) {
  const int lane = threadIdx.x & 31, wave = threadIdx.x >> 5;
  const int m0 = blockIdx.y * 256 + wave * 32;
  const int n0 = blockIdx.x * 64;
  Acc8 acc = gemm_wave_32x64(A, BT, K, K, K, m0, n0, lane);
  const int lm = lane & 15, h8 = (lane >= 16) ? 8 : 0;
#pragma unroll
  for (int t = 0; t < 4; ++t)
#pragma unroll
    for (int j = 0; j < 8; ++j) {
      C[(size_t)(m0 + j + h8) * N + (n0 + t * 16 + lm)]      = (__bf16)(acc.r0[t][j]);
      C[(size_t)(m0 + 16 + j + h8) * N + (n0 + t * 16 + lm)] = (__bf16)(acc.r1[t][j]);
    }
}

// Output projection GEMM -> fp32 + bias (final result)
__global__ void gemm_out_kernel(const __bf16* __restrict__ A,
                                const __bf16* __restrict__ BT,
                                const float* __restrict__ bias,
                                float* __restrict__ C) {
  const int lane = threadIdx.x & 31, wave = threadIdx.x >> 5;
  const int m0 = blockIdx.y * 256 + wave * 32;
  const int n0 = blockIdx.x * 64;
  Acc8 acc = gemm_wave_32x64(A, BT, 1024, 1024, 1024, m0, n0, lane);
  const int lm = lane & 15, h8 = (lane >= 16) ? 8 : 0;
#pragma unroll
  for (int t = 0; t < 4; ++t) {
    const int n = n0 + t * 16 + lm;
    const float bv = bias[n];
#pragma unroll
    for (int j = 0; j < 8; ++j) {
      C[(size_t)(m0 + j + h8) * 1024 + n]      = acc.r0[t][j] + bv;
      C[(size_t)(m0 + 16 + j + h8) * 1024 + n] = acc.r1[t][j] + bv;
    }
  }
}

// ---------------------------------------------------------------------------
// KV projection GEMM with TDM-staged B.  A = kvn (32768 x 1024),
// BT = w_kvT (2048 x 1024).  Each 256x64 block stages its 64-row B slab
// through LDS in four 64x256-bf16 chunks (32 KB each, double buffered) moved
// by tensor_load_to_lds; all 8 waves share each chunk.  Epilogue scatters
// K -> [b,h,n2,64] and V -> TRANSPOSED [b,h,64,n2].
// ---------------------------------------------------------------------------
__global__ void gemm_kv_kernel(const __bf16* __restrict__ A,
                               const __bf16* __restrict__ BT,
                               __bf16* __restrict__ Kb,
                               __bf16* __restrict__ VT) {
  extern __shared__ __bf16 ldsB[];   // 2 x 64 x 256 bf16 = 64 KB
  const int lane = threadIdx.x & 31, wave = threadIdx.x >> 5;
  const int lm = lane & 15, h8 = (lane >= 16) ? 8 : 0;
  const int m0 = blockIdx.y * 256 + wave * 32;
  const int n0 = blockIdx.x * 64;

  const __bf16* arow0 = A + (size_t)(m0 + lm) * 1024;
  const __bf16* arow1 = arow0 + (size_t)16 * 1024;

  const unsigned ldsBase = (unsigned)(uintptr_t)ldsB;  // low 32 bits = LDS byte offset
  const unsigned long long gbase =
      (unsigned long long)(uintptr_t)(BT + (size_t)n0 * 1024);

  v8f acc0[4] = {}, acc1[4] = {};

  // Prime chunk 0: 64 rows x 256 cols of the slab-tensor (1024 wide, stride 1024)
  if (wave == 0) {
    tdm_load_2d(ldsBase, gbase, /*td0=*/1024, /*td1=*/64,
                /*tile0=*/256, /*tile1=*/64, /*stride0=*/1024);
    __builtin_amdgcn_s_wait_tensorcnt(0);
  }
  __syncthreads();

  for (int c = 0; c < 4; ++c) {
    if (wave == 0 && c < 3)
      tdm_load_2d(ldsBase + (unsigned)(((c + 1) & 1) * 32768),
                  gbase + (unsigned long long)(c + 1) * 256 * 2,
                  1024, 64, 256, 64, 1024);
    const __bf16* bbuf = ldsB + (size_t)(c & 1) * 16384;  // elements
    const int kc = c * 256;
#pragma unroll
    for (int kb = 0; kb < 256; kb += 32) {
      v16bf a0 = load_frag(arow0, kc + kb, lane);
      v16bf a1 = load_frag(arow1, kc + kb, lane);
#pragma unroll
      for (int t = 0; t < 4; ++t) {
        v16bf bt = load_frag(bbuf + (size_t)(t * 16 + lm) * 256, kb, lane);
        acc0[t] = wmma_bf16(a0, bt, acc0[t]);
        acc1[t] = wmma_bf16(a1, bt, acc1[t]);
      }
    }
    if (wave == 0 && c < 3) __builtin_amdgcn_s_wait_tensorcnt(0);
    __syncthreads();
  }

#pragma unroll
  for (int t = 0; t < 4; ++t)
#pragma unroll
    for (int j = 0; j < 8; ++j)
#pragma unroll
      for (int r = 0; r < 2; ++r) {
        const int m  = m0 + r * 16 + j + h8;
        const int b  = m >> 12, n2 = m & 4095;
        const int cc = n0 + t * 16 + lm;
        const float val = r ? acc1[t][j] : acc0[t][j];
        if (cc < 1024) {
          const int h = cc >> 6, d = cc & 63;
          Kb[(((size_t)(b * 16 + h) * 4096) + n2) * 64 + d] = (__bf16)val;
        } else {
          const int c2 = cc - 1024, h = c2 >> 6, d = c2 & 63;
          VT[(((size_t)(b * 16 + h) * 64) + d) * 4096 + n2] = (__bf16)val;
        }
      }
}

// ---------------------------------------------------------------------------
// Flash attention: one wave = one 16-row Q tile of one (b,h); streams N2=4096
// in 32-wide chunks.  Q: [2048][1024] bf16.  K: [b,h,4096,64].  VT: [b,h,64,4096].
// ---------------------------------------------------------------------------
__global__ void attention_kernel(const __bf16* __restrict__ Q,
                                 const __bf16* __restrict__ Kb,
                                 const __bf16* __restrict__ VT,
                                 const int* __restrict__ mask,
                                 __bf16* __restrict__ O) {
  __shared__ __attribute__((aligned(16))) __bf16 plds[4][16][32];
  const int lane = threadIdx.x & 31, wave = threadIdx.x >> 5;
  const int lm = lane & 15, h8 = (lane >= 16) ? 8 : 0;
  const int tile = blockIdx.x * 4 + wave;     // 2048 tiles total
  const int bh = tile >> 4, qt = tile & 15;
  const int b = bh >> 4, h = bh & 15;

  const __bf16* qrow = Q + (size_t)(b * 256 + qt * 16 + lm) * 1024 + h * 64;
  const v16bf qa0 = load_frag(qrow, 0, lane);
  const v16bf qa1 = load_frag(qrow, 32, lane);
  const __bf16* kbase = Kb + (size_t)bh * 4096 * 64;
  const __bf16* vbase = VT + (size_t)bh * 64 * 4096;
  const int* mp = mask + b * 4096;

  float mrun[8], lrun[8];
  v8f o0 = {}, o1 = {}, o2 = {}, o3 = {};
#pragma unroll
  for (int j = 0; j < 8; ++j) { mrun[j] = -__builtin_inff(); lrun[j] = 0.f; }

  for (int n2 = 0; n2 < 4096; n2 += 32) {
    // ---- S = Q @ K^T (16 x 32 chunk), B-fragment rows = K rows (contiguous)
    const __bf16* kr0 = kbase + (size_t)(n2 + lm) * 64;
    const __bf16* kr1 = kr0 + 16 * 64;
    v8f s0 = {}, s1 = {};
    s0 = wmma_bf16(qa0, load_frag(kr0, 0,  lane), s0);
    s0 = wmma_bf16(qa1, load_frag(kr0, 32, lane), s0);
    s1 = wmma_bf16(qa0, load_frag(kr1, 0,  lane), s1);
    s1 = wmma_bf16(qa1, load_frag(kr1, 32, lane), s1);

    const int mv0 = mp[n2 + lm], mv1 = mp[n2 + 16 + lm];
#pragma unroll
    for (int j = 0; j < 8; ++j) {
      float a0 = mv0 ? s0[j] * ATT_SCALE : MASK_FILL;
      float a1 = mv1 ? s1[j] * ATT_SCALE : MASK_FILL;
      // row statistics: each row lives across a 16-lane group (xor 1,2,4,8)
      float vmax = fmaxf(a0, a1);
#pragma unroll
      for (int off = 8; off; off >>= 1) vmax = fmaxf(vmax, __shfl_xor(vmax, off, 32));
      const float mnew  = fmaxf(mrun[j], vmax);
      const float alpha = __expf(mrun[j] - mnew);
      mrun[j] = mnew;
      const float p0 = __expf(a0 - mnew), p1 = __expf(a1 - mnew);
      float rs = p0 + p1;
#pragma unroll
      for (int off = 8; off; off >>= 1) rs += __shfl_xor(rs, off, 32);
      lrun[j] = lrun[j] * alpha + rs;
      o0[j] *= alpha; o1[j] *= alpha; o2[j] *= alpha; o3[j] *= alpha;
      // C-layout -> A-layout re-swizzle through per-wave LDS (in-order per wave)
      plds[wave][j + h8][lm]      = (__bf16)p0;
      plds[wave][j + h8][16 + lm] = (__bf16)p1;
    }

    // ---- O += P @ V (B-fragment rows = VT rows over n2: contiguous)
    const v16bf pa = load_frag(&plds[wave][lm][0], 0, lane);
    o0 = wmma_bf16(pa, load_frag(vbase + (size_t)(lm)      * 4096 + n2, 0, lane), o0);
    o1 = wmma_bf16(pa, load_frag(vbase + (size_t)(16 + lm) * 4096 + n2, 0, lane), o1);
    o2 = wmma_bf16(pa, load_frag(vbase + (size_t)(32 + lm) * 4096 + n2, 0, lane), o2);
    o3 = wmma_bf16(pa, load_frag(vbase + (size_t)(48 + lm) * 4096 + n2, 0, lane), o3);
  }

  const v8f* os[4] = {&o0, &o1, &o2, &o3};
#pragma unroll
  for (int t = 0; t < 4; ++t)
#pragma unroll
    for (int j = 0; j < 8; ++j) {
      const float val = (*os[t])[j] / lrun[j];
      O[(size_t)(b * 256 + qt * 16 + j + h8) * 1024 + h * 64 + t * 16 + lm] = (__bf16)val;
    }
}

// ---------------------------------------------------------------------------
// Host-side orchestration
// ---------------------------------------------------------------------------
extern "C" void kernel_launch(void* const* d_in, const int* in_sizes, int n_in,
                              void* d_out, int out_size, void* d_ws, size_t ws_size,
                              hipStream_t stream) {
  const float* x      = (const float*)d_in[0];   // (8, 256, 1024)
  const float* kv     = (const float*)d_in[1];   // (8, 4096, 1024)
  const int*   pmask  = (const int*)  d_in[2];   // (8, 4096)
  const float* w_q    = (const float*)d_in[3];   // (1024, 1024)
  const float* w_kv   = (const float*)d_in[4];   // (1024, 2048)
  const float* w_proj = (const float*)d_in[5];   // (1024, 1024)
  const float* b_proj = (const float*)d_in[6];   // (1024,)
  const float* g_x    = (const float*)d_in[7];
  const float* b_x    = (const float*)d_in[8];
  const float* g_kv   = (const float*)d_in[9];
  const float* b_kv   = (const float*)d_in[10];
  float* out = (float*)d_out;

  // Workspace partition (all 16B aligned; total ~212 MiB)
  char* w = (char*)d_ws;
  __bf16* xn     = (__bf16*)w; w += (size_t)2048  * 1024 * 2;
  __bf16* kvn    = (__bf16*)w; w += (size_t)32768 * 1024 * 2;
  __bf16* wqT    = (__bf16*)w; w += (size_t)1024  * 1024 * 2;
  __bf16* wkvT   = (__bf16*)w; w += (size_t)2048  * 1024 * 2;
  __bf16* wprojT = (__bf16*)w; w += (size_t)1024  * 1024 * 2;
  __bf16* qb     = (__bf16*)w; w += (size_t)2048  * 1024 * 2;
  __bf16* kb     = (__bf16*)w; w += (size_t)128 * 4096 * 64 * 2;
  __bf16* vt     = (__bf16*)w; w += (size_t)128 * 4096 * 64 * 2;
  __bf16* att    = (__bf16*)w; w += (size_t)2048  * 1024 * 2;

  // 1) LayerNorm + bf16 convert
  layernorm_to_bf16<<<2048,  256, 0, stream>>>(x,  g_x,  b_x,  xn);
  layernorm_to_bf16<<<32768, 256, 0, stream>>>(kv, g_kv, b_kv, kvn);

  // 2) Transpose weights to N-major bf16 (B-fragment loads become contiguous)
  transpose_to_bf16<<<dim3(32, 32), dim3(32, 8), 0, stream>>>(w_q,    wqT,    1024, 1024);
  transpose_to_bf16<<<dim3(64, 32), dim3(32, 8), 0, stream>>>(w_kv,   wkvT,   1024, 2048);
  transpose_to_bf16<<<dim3(32, 32), dim3(32, 8), 0, stream>>>(w_proj, wprojT, 1024, 1024);

  // 3) Q projection: (2048 x 1024) = xn @ w_q
  gemm_bf16_kernel<<<dim3(16, 8), 256, 0, stream>>>(xn, wqT, qb, 1024, 1024);

  // 4) KV projection (TDM-staged B): (32768 x 2048) = kvn @ w_kv
  gemm_kv_kernel<<<dim3(32, 128), 256, 65536, stream>>>(kvn, wkvT, kb, vt);

  // 5) Flash attention, one wave per 16-row Q tile (2048 tiles, 4 waves/block)
  attention_kernel<<<512, 128, 0, stream>>>(qb, kb, vt, pmask, att);

  // 6) Output projection + bias -> fp32 d_out
  gemm_out_kernel<<<dim3(16, 8), 256, 0, stream>>>(att, wprojT, b_proj, out);

  (void)in_sizes; (void)n_in; (void)out_size; (void)ws_size;
}
